// SELayer_89979564851291
// MI455X (gfx1250) — matrix operand order repaired and checked by reading, not compiled, not run on hardware
//
#include <hip/hip_runtime.h>
#include <cstddef>

// ---------------------------------------------------------------------------
// CDNA5 (gfx1250) DiT block: adaLN + attention + MLP, all GEMMs on
// v_wmma_f32_16x16x32_bf16 (wave32). Flash-attention style fused softmax.
// Async global->LDS staging (ASYNCcnt) via global_load_async_to_lds_b128.
// ---------------------------------------------------------------------------

typedef __attribute__((ext_vector_type(16))) __bf16 v16bf;
typedef __attribute__((ext_vector_type(8)))  float  v8f;
typedef __attribute__((ext_vector_type(4)))  int    v4i;

#define WMMA_BF16(a, b, c) \
  __builtin_amdgcn_wmma_f32_16x16x32_bf16(false, (a), false, (b), (short)0, (c), false, false)

#if defined(__HIP_DEVICE_COMPILE__) && __has_builtin(__builtin_amdgcn_global_load_async_to_lds_b128)
#define HAVE_ASYNC_LDS 1
#else
#define HAVE_ASYNC_LDS 0
#endif

static __device__ __forceinline__ void wait_async0() {
#if HAVE_ASYNC_LDS
#if __has_builtin(__builtin_amdgcn_s_wait_asynccnt)
  __builtin_amdgcn_s_wait_asynccnt(0);
#else
  asm volatile("s_wait_asynccnt 0x0" ::: "memory");
#endif
#endif
}

// 128-bit copy global -> LDS (8 bf16 per call)
static __device__ __forceinline__ void cp_b128(const unsigned short* src, unsigned short* dst) {
#if HAVE_ASYNC_LDS
  v4i* s0 = (v4i*)src;  // C-style cast drops const
  v4i* d0 = (v4i*)dst;
  __builtin_amdgcn_global_load_async_to_lds_b128(
      (__attribute__((address_space(1))) v4i*)s0,
      (__attribute__((address_space(3))) v4i*)d0, 0, 0);
#else
  *(uint4*)dst = *(const uint4*)src;
#endif
}

static __device__ __forceinline__ unsigned short f2bf(float f) {
  union { __bf16 b; unsigned short s; } v;
  v.b = (__bf16)f;                                  // native v_cvt_*bf16_f32
  return v.s;
}

// A-matrix 16x32 bf16 fragment (ISA 7.12.2): lane<16 -> row=lr, K = k0..k0+7 and
// k0+16..k0+23 ; lane>=16 -> K = k0+8..15 and k0+24..31.
static __device__ __forceinline__ v16bf load_fragA(const unsigned short* base, int ldk,
                                                   int row, int k0, int hf) {
  const unsigned short* p = base + (size_t)row * ldk + k0 + hf * 8;
  union { uint4 u[2]; v16bf v; } c;
  c.u[0] = *(const uint4*)p;
  c.u[1] = *(const uint4*)(p + 16);
  return c.v;
}
// B-matrix 32x16 bf16 fragment: lane<16 -> col=lr, K = k0..k0+7 and k0+16..k0+23;
// lane>=16 -> K = k0+8..15 and k0+24..31. base is B^T stored [N][K].
static __device__ __forceinline__ v16bf load_fragB(const unsigned short* base, int ldk,
                                                   int row, int k0, int hf) {
  const unsigned short* p = base + (size_t)row * ldk + k0 + hf * 8;
  union { uint4 u[2]; v16bf v; } c;
  c.u[0] = *(const uint4*)p;
  c.u[1] = *(const uint4*)(p + 16);
  return c.v;
}

// ---------------------------------------------------------------------------
// Weight prep: fp32 W[K][N] -> bf16 W^T[N][K]
// ---------------------------------------------------------------------------
__global__ void w_to_bf16_t(const float* __restrict__ W, unsigned short* __restrict__ Wt,
                            int K, int N) {
  int i = blockIdx.x * 256 + threadIdx.x;          // i = n*K + k
  int n = i / K, k = i - n * K;
  Wt[i] = f2bf(W[(size_t)k * N + n]);
}

// ---------------------------------------------------------------------------
// adaLN (tiny): h = silu(c @ W1 + b1) ; ada = h @ W2 + b2
// ---------------------------------------------------------------------------
__global__ void ada_stage1(const float* __restrict__ c, const float* __restrict__ W,
                           const float* __restrict__ bvec, float* __restrict__ h) {
  int i = blockIdx.x * 256 + threadIdx.x;          // 4*1024
  int b = i >> 10, j = i & 1023;
  float acc = 0.f;
  for (int kk = 0; kk < 1024; ++kk) acc += c[b * 1024 + kk] * W[(size_t)kk * 1024 + j];
  acc += bvec[j];
  h[i] = acc / (1.f + expf(-acc));                 // silu: x*sigmoid(x)
}
__global__ void ada_stage2(const float* __restrict__ h, const float* __restrict__ W,
                           const float* __restrict__ bvec, float* __restrict__ ada) {
  int i = blockIdx.x * 256 + threadIdx.x;          // 4*6144
  int b = i / 6144, j = i - b * 6144;
  float acc = 0.f;
  for (int kk = 0; kk < 1024; ++kk) acc += h[b * 1024 + kk] * W[(size_t)kk * 6144 + j];
  ada[i] = acc + bvec[j];
}

// ---------------------------------------------------------------------------
// Generic WMMA GEMM: C[M][N] = A_fp32[M][K] * B (B^T bf16 [N][K]) (+bias)(+gelu)
// 128x128 tile, TK=32, 256 threads = 8 waves, wave does 32x64 (2x4 frags).
// Double-buffered LDS; B staged via GLOBAL_LOAD_ASYNC_TO_LDS.
// ---------------------------------------------------------------------------
enum { EPI_NONE = 0, EPI_BIAS = 1, EPI_BIAS_GELU = 2 };

__global__ __launch_bounds__(256) void gemm_a32_bt16(
    const float* __restrict__ A, const unsigned short* __restrict__ Bt,
    const float* __restrict__ bias, float* __restrict__ C,
    int M, int N, int K, int epi) {
  __shared__ unsigned short As[2][128 * 32];
  __shared__ unsigned short Bs[2][128 * 32];
  const int tid = threadIdx.x;
  const int wave = tid >> 5, lane = tid & 31, hf = lane >> 4, lr = lane & 15;
  const int wm = (wave & 3) * 32, wn = (wave >> 2) * 64;
  const int m0 = blockIdx.y * 128, n0 = blockIdx.x * 128;

  v8f acc[2][4];
  for (int i = 0; i < 2; ++i)
    for (int j = 0; j < 4; ++j)
      for (int r = 0; r < 8; ++r) acc[i][j][r] = 0.f;

  const int srow = tid >> 1, sseg = tid & 1;
  const int nk = K >> 5;

  auto stageB = [&](int buf, int kt) {
    const unsigned short* bp = Bt + (size_t)(n0 + srow) * K + kt + sseg * 16;
    unsigned short* dp = &Bs[buf][srow * 32 + sseg * 16];
    cp_b128(bp, dp);
    cp_b128(bp + 8, dp + 8);
  };
  auto stageA = [&](int buf, int kt) {
    const float* ap = A + (size_t)(m0 + srow) * K + kt + sseg * 16;
    unsigned short* dp = &As[buf][srow * 32 + sseg * 16];
#pragma unroll
    for (int v = 0; v < 4; ++v) {
      float4 f = ((const float4*)ap)[v];
      union { __bf16 b[4]; uint2 u; } pk;
      pk.b[0] = (__bf16)f.x; pk.b[1] = (__bf16)f.y;
      pk.b[2] = (__bf16)f.z; pk.b[3] = (__bf16)f.w;
      ((uint2*)dp)[v] = pk.u;
    }
    if (kt + 32 < K) __builtin_prefetch(ap + 32, 0, 1);  // global_prefetch_b8
  };

  stageB(0, 0);
  stageA(0, 0);
  wait_async0();
  __syncthreads();

  for (int it = 0; it < nk; ++it) {
    const int cur = it & 1;
    if (it + 1 < nk) {          // issue next tile; overlaps with WMMA below
      stageB(cur ^ 1, (it + 1) * 32);
      stageA(cur ^ 1, (it + 1) * 32);
    }
    v16bf a0 = load_fragA(&As[cur][0], 32, wm + lr, 0, hf);
    v16bf a1 = load_fragA(&As[cur][0], 32, wm + 16 + lr, 0, hf);
#pragma unroll
    for (int nf = 0; nf < 4; ++nf) {
      v16bf bb = load_fragB(&Bs[cur][0], 32, wn + nf * 16 + lr, 0, hf);
      acc[0][nf] = WMMA_BF16(a0, bb, acc[0][nf]);
      acc[1][nf] = WMMA_BF16(a1, bb, acc[1][nf]);
    }
    wait_async0();
    __syncthreads();
  }
#pragma unroll
  for (int mf = 0; mf < 2; ++mf)
#pragma unroll
    for (int nf = 0; nf < 4; ++nf)
#pragma unroll
      for (int r = 0; r < 8; ++r) {
        int row = m0 + wm + mf * 16 + r + hf * 8;
        int col = n0 + wn + nf * 16 + lr;
        float v = acc[mf][nf][r];
        if (epi >= EPI_BIAS) v += bias[col];
        if (epi == EPI_BIAS_GELU) v = 0.5f * v * (1.f + erff(v * 0.70710678118f));
        C[(size_t)row * N + col] = v;
      }
}

// ---------------------------------------------------------------------------
// split qkv fp32 [B*N][3C] -> q,k bf16 [B][H][N][64] (q pre-scaled D^-1/2),
// v bf16 transposed [B][H][64][N]
// ---------------------------------------------------------------------------
__global__ void split_qkv(const float* __restrict__ qkv, unsigned short* __restrict__ Qo,
                          unsigned short* __restrict__ Ko, unsigned short* __restrict__ VTo) {
  int i = blockIdx.x * 256 + threadIdx.x;  // B*N*H*D = 4M
  int d = i & 63, h = (i >> 6) & 15, n = (i >> 10) & 1023, b = i >> 20;
  size_t base = ((size_t)(b * 1024 + n)) * 3072 + h * 64 + d;
  float qv = qkv[base], kv = qkv[base + 1024], vv = qkv[base + 2048];
  size_t bh = (size_t)(b * 16 + h);
  Qo[(bh * 1024 + n) * 64 + d] = f2bf(qv * 0.125f);  // 1/sqrt(64)
  Ko[(bh * 1024 + n) * 64 + d] = f2bf(kv);
  VTo[(bh * 64 + d) * 1024 + n] = f2bf(vv);
}

// ---------------------------------------------------------------------------
// Flash attention: per block (b,h, 128 query rows); 8 key blocks of 128.
// LDS: Ks 16KB | Vs 16KB | P-extra 16KB (P for waves 0-3 reuses Ks region)
// ---------------------------------------------------------------------------
__global__ __launch_bounds__(256) void flash_attn_kernel(
    const unsigned short* __restrict__ Q, const unsigned short* __restrict__ Kt,
    const unsigned short* __restrict__ VT, const unsigned char* __restrict__ mask,
    float* __restrict__ out) {
  __shared__ unsigned short sm[24832];     // 49664 B
  float* mf = (float*)(sm + 24576);        // 128 mask floats
  const int tid = threadIdx.x;
  const int wave = tid >> 5, lane = tid & 31, hf = lane >> 4, lr = lane & 15;
  const int b = blockIdx.z, h = blockIdx.y, rt = blockIdx.x;
  const size_t bh = (size_t)(b * 16 + h);

  const unsigned short* qbase = Q + (bh * 1024 + rt * 128 + wave * 16) * 64;
  v16bf aq0 = load_fragA(qbase, 64, lr, 0, hf);
  v16bf aq1 = load_fragA(qbase, 64, lr, 32, hf);

  v8f o[4];
  float mrow[8], lrow[8];
  for (int j = 0; j < 4; ++j)
    for (int r = 0; r < 8; ++r) o[j][r] = 0.f;
  for (int r = 0; r < 8; ++r) { mrow[r] = -1e30f; lrow[r] = 0.f; }

  unsigned short* pw = (wave < 4) ? (sm + wave * 2048) : (sm + 16384 + (wave - 4) * 2048);

  for (int kb = 0; kb < 8; ++kb) {
    __syncthreads();
    {  // stage K block [128 keys][64] bf16 (async -> LDS)
      int row = tid >> 1, seg = tid & 1;
      const unsigned short* src = Kt + (bh * 1024 + kb * 128 + row) * 64 + seg * 32;
      unsigned short* dst = sm + row * 64 + seg * 32;
#pragma unroll
      for (int v = 0; v < 4; ++v) cp_b128(src + v * 8, dst + v * 8);
    }
    {  // stage V^T block [64 d][128 keys] bf16 (async -> LDS)
      int row = tid >> 2, q4 = tid & 3;
      const unsigned short* src = VT + (bh * 64 + row) * 1024 + kb * 128 + q4 * 32;
      unsigned short* dst = sm + 8192 + row * 128 + q4 * 32;
#pragma unroll
      for (int v = 0; v < 4; ++v) cp_b128(src + v * 8, dst + v * 8);
    }
    if (tid < 128) mf[tid] = mask[b * 1024 + kb * 128 + tid] ? 0.f : -1e30f;
    wait_async0();
    __syncthreads();

    // S = q @ k^T  (16 x 128 per wave)
    v8f s[8];
#pragma unroll
    for (int nf = 0; nf < 8; ++nf) {
      for (int r = 0; r < 8; ++r) s[nf][r] = 0.f;
      v16bf b0 = load_fragB(sm, 64, nf * 16 + lr, 0, hf);
      s[nf] = WMMA_BF16(aq0, b0, s[nf]);
      v16bf b1 = load_fragB(sm, 64, nf * 16 + lr, 32, hf);
      s[nf] = WMMA_BF16(aq1, b1, s[nf]);
    }
    // key mask (column = nf*16 + lr)
#pragma unroll
    for (int nf = 0; nf < 8; ++nf) {
      float ma = mf[nf * 16 + lr];
      for (int r = 0; r < 8; ++r) s[nf][r] += ma;
    }
    // online softmax: rows live in half-wave groups of 16 lanes
    float sc[8];
#pragma unroll
    for (int r = 0; r < 8; ++r) {
      float x = s[0][r];
      for (int nf = 1; nf < 8; ++nf) x = fmaxf(x, s[nf][r]);
      x = fmaxf(x, __shfl_xor(x, 1));
      x = fmaxf(x, __shfl_xor(x, 2));
      x = fmaxf(x, __shfl_xor(x, 4));
      x = fmaxf(x, __shfl_xor(x, 8));
      float mn = fmaxf(mrow[r], x);
      sc[r] = expf(mrow[r] - mn);
      mrow[r] = mn;
    }
    __syncthreads();  // all waves done reading Ks before P overwrites it
#pragma unroll
    for (int r = 0; r < 8; ++r) {
      float rs = 0.f;
      for (int nf = 0; nf < 8; ++nf) {
        float p = expf(s[nf][r] - mrow[r]);
        rs += p;
        pw[(r + hf * 8) * 128 + nf * 16 + lr] = f2bf(p);
      }
      rs += __shfl_xor(rs, 1);
      rs += __shfl_xor(rs, 2);
      rs += __shfl_xor(rs, 4);
      rs += __shfl_xor(rs, 8);
      lrow[r] = lrow[r] * sc[r] + rs;
    }
#pragma unroll
    for (int j = 0; j < 4; ++j)
      for (int r = 0; r < 8; ++r) o[j][r] *= sc[r];
    __syncthreads();
    // O += P @ V
#pragma unroll
    for (int af = 0; af < 4; ++af) {
      v16bf pa = load_fragA(pw, 128, lr, af * 32, hf);
#pragma unroll
      for (int j = 0; j < 4; ++j) {
        v16bf bv = load_fragB(sm + 8192, 128, j * 16 + lr, af * 32, hf);
        o[j] = WMMA_BF16(pa, bv, o[j]);
      }
    }
  }
  // write attn_out fp32 [B][N][C], col = h*64 + d
#pragma unroll
  for (int j = 0; j < 4; ++j)
#pragma unroll
    for (int r = 0; r < 8; ++r) {
      int row = rt * 128 + wave * 16 + r + hf * 8;
      int col = h * 64 + j * 16 + lr;
      out[((size_t)b * 1024 + row) * 1024 + col] = o[j][r] / lrow[r];
    }
}

// ---------------------------------------------------------------------------
// Fused LayerNorm + modulate + gate + residual: one block per (b,n) row.
// out = xin + gate * ( LN(y) * (1+scale) + shift )
// ---------------------------------------------------------------------------
__global__ __launch_bounds__(256) void ln_mod_res(
    const float* __restrict__ xin, const float* __restrict__ y,
    const float* __restrict__ ada, int sh_off, int sc_off, int g_off,
    float* __restrict__ out) {
  __shared__ float rs[256], rq[256];
  const int tid = threadIdx.x;
  const int b = blockIdx.x >> 10;
  const size_t base = (size_t)blockIdx.x * 1024;
  float s = 0.f, q = 0.f, vals[4];
#pragma unroll
  for (int i = 0; i < 4; ++i) {
    float v = y[base + tid + i * 256];
    vals[i] = v; s += v; q += v * v;
  }
  rs[tid] = s; rq[tid] = q;
  __syncthreads();
  for (int st = 128; st > 0; st >>= 1) {
    if (tid < st) { rs[tid] += rs[tid + st]; rq[tid] += rq[tid + st]; }
    __syncthreads();
  }
  float mu = rs[0] * (1.f / 1024.f);
  float var = rq[0] * (1.f / 1024.f) - mu * mu;
  float rstd = rsqrtf(var + 1e-5f);
  const float* ab = ada + (size_t)b * 6144;
#pragma unroll
  for (int i = 0; i < 4; ++i) {
    int cidx = tid + i * 256;
    float nv = (vals[i] - mu) * rstd;
    float g = ab[g_off + cidx], scv = ab[sc_off + cidx], shv = ab[sh_off + cidx];
    out[base + cidx] = xin[base + cidx] + g * (nv * (1.f + scv) + shv);
  }
}

// ---------------------------------------------------------------------------
extern "C" void kernel_launch(void* const* d_in, const int* in_sizes, int n_in,
                              void* d_out, int out_size, void* d_ws, size_t ws_size,
                              hipStream_t stream) {
  (void)in_sizes; (void)n_in; (void)out_size; (void)ws_size;
  const float* x       = (const float*)d_in[0];
  const float* c       = (const float*)d_in[1];
  const unsigned char* node_mask = (const unsigned char*)d_in[2];  // jnp bool = 1 byte
  const float* qkv_w   = (const float*)d_in[3];
  const float* proj_w  = (const float*)d_in[4];
  const float* proj_b  = (const float*)d_in[5];
  const float* fc1_w   = (const float*)d_in[6];
  const float* fc1_b   = (const float*)d_in[7];
  const float* fc2_w   = (const float*)d_in[8];
  const float* fc2_b   = (const float*)d_in[9];
  const float* ada1_w  = (const float*)d_in[10];
  const float* ada1_b  = (const float*)d_in[11];
  const float* ada2_w  = (const float*)d_in[12];
  const float* ada2_b  = (const float*)d_in[13];
  float* out = (float*)d_out;

  char* wsb = (char*)d_ws;
  size_t off = 0;
  auto alloc = [&](size_t bytes) -> char* {
    char* p = wsb + off;
    off += (bytes + 255) & ~(size_t)255;
    return p;
  };
  unsigned short* qkvT  = (unsigned short*)alloc(3072ull * 1024 * 2);
  unsigned short* projT = (unsigned short*)alloc(1024ull * 1024 * 2);
  unsigned short* fc1T  = (unsigned short*)alloc(4096ull * 1024 * 2);
  unsigned short* fc2T  = (unsigned short*)alloc(1024ull * 4096 * 2);
  float* adaH  = (float*)alloc(4ull * 1024 * 4);
  float* adaO  = (float*)alloc(4ull * 6144 * 4);
  float* qkvO  = (float*)alloc(4096ull * 3072 * 4);
  unsigned short* Qb  = (unsigned short*)alloc(4ull * 16 * 1024 * 64 * 2);
  unsigned short* Kb  = (unsigned short*)alloc(4ull * 16 * 1024 * 64 * 2);
  unsigned short* VTb = (unsigned short*)alloc(4ull * 16 * 64 * 1024 * 2);
  float* attnO = (float*)alloc(4096ull * 1024 * 4);
  float* projO = (float*)alloc(4096ull * 1024 * 4);
  float* x1    = (float*)alloc(4096ull * 1024 * 4);
  float* hMLP  = (float*)alloc(4096ull * 4096 * 4);
  float* mlpO  = (float*)alloc(4096ull * 1024 * 4);

  // weight prep (bf16, transposed)
  w_to_bf16_t<<<3072 * 1024 / 256, 256, 0, stream>>>(qkv_w, qkvT, 1024, 3072);
  w_to_bf16_t<<<1024 * 1024 / 256, 256, 0, stream>>>(proj_w, projT, 1024, 1024);
  w_to_bf16_t<<<4096 * 1024 / 256, 256, 0, stream>>>(fc1_w, fc1T, 1024, 4096);
  w_to_bf16_t<<<1024 * 4096 / 256, 256, 0, stream>>>(fc2_w, fc2T, 4096, 1024);

  // adaLN
  ada_stage1<<<16, 256, 0, stream>>>(c, ada1_w, ada1_b, adaH);
  ada_stage2<<<96, 256, 0, stream>>>(adaH, ada2_w, ada2_b, adaO);

  // qkv = x @ qkv_w
  gemm_a32_bt16<<<dim3(24, 32), 256, 0, stream>>>(x, qkvT, nullptr, qkvO,
                                                  4096, 3072, 1024, EPI_NONE);
  split_qkv<<<16384, 256, 0, stream>>>(qkvO, Qb, Kb, VTb);

  // fused attention
  flash_attn_kernel<<<dim3(8, 16, 4), 256, 0, stream>>>(Qb, Kb, VTb, node_mask, attnO);

  // proj
  gemm_a32_bt16<<<dim3(8, 32), 256, 0, stream>>>(attnO, projT, proj_b, projO,
                                                 4096, 1024, 1024, EPI_BIAS);
  // x1 = x + gate_msa * modulate(LN(proj_out))
  ln_mod_res<<<4096, 256, 0, stream>>>(x, projO, adaO, 0, 1024, 2048, x1);

  // MLP
  gemm_a32_bt16<<<dim3(32, 32), 256, 0, stream>>>(x1, fc1T, fc1_b, hMLP,
                                                  4096, 4096, 1024, EPI_BIAS_GELU);
  gemm_a32_bt16<<<dim3(8, 32), 256, 0, stream>>>(hMLP, fc2T, fc2_b, mlpO,
                                                 4096, 1024, 4096, EPI_BIAS);
  // out = x1 + gate_mlp * modulate(LN(mlp_out))
  ln_mod_res<<<4096, 256, 0, stream>>>(x1, mlpO, adaO, 3072, 4096, 5120, out);
}
